// TransformerBlock_47038481826405
// MI455X (gfx1250) — compile-verified, matrix operand order checked by reading, not run on hardware
//
#include <hip/hip_runtime.h>
#include <hip/hip_bf16.h>
#include <math.h>

typedef __attribute__((ext_vector_type(16))) __bf16 v16bf;
typedef __attribute__((ext_vector_type(8)))  __bf16 v8bf;
typedef __attribute__((ext_vector_type(8)))  float  v8f;

constexpr int kC     = 192;   // channels
constexpr int kN     = 64;    // tokens per window (8x8)
constexpr int kHeads = 6;
constexpr int kD     = 32;    // head dim
constexpr int kMLP   = 768;
constexpr int kWS    = 8;
constexpr int kIMG   = 256;
constexpr int kB     = 4;
constexpr int kNWin  = kB * (kIMG / kWS) * (kIMG / kWS);  // 4096

// LDS strides (bf16 elements). All multiples of 8 -> 16B-aligned b128 frag loads.
constexpr int SA   = 200;  // A-staging (LN out)
constexpr int SQK  = 200;  // q, k (token-major)
constexpr int SVT  = 72;   // vT (channel-major)
constexpr int SSC  = 72;   // scores row stride (64 + pad)
constexpr int SHID = 392;  // MLP hidden chunk (384 + pad)

struct Smem {
  float t[kN][kC];                                   // 48 KB residual (f32)
  union {
    __bf16 a[kN][SA];                                // LN output, A matrix staging
    __bf16 sc[kHeads][kN][SSC];                      // attention scores / probs
  } A;                                               // 55.3 KB
  union {
    struct { __bf16 q[kN][SQK]; __bf16 k[kN][SQK]; __bf16 vT[kC][SVT]; } qkv;
    __bf16 hid[kN][SHID];                            // MLP hidden chunk
  } B;                                               // 78.8 KB
};                                                   // ~179 KB total (< 320KB/WGP)

__device__ __forceinline__ v8f wmma_bf16(v16bf a, v16bf b, v8f c) {
  // v_wmma_f32_16x16x32_bf16: D = A(16x32) x B(32x16) + C
  return __builtin_amdgcn_wmma_f32_16x16x32_bf16(
      false, a, false, b, (short)0, c, false, false);
}

// Load a 16x32 bf16 A/B fragment from a row-major buffer (row = M for A, N for B).
// ISA 16-bit layout: lanes 0-15 hold K {0..7,16..23}, lanes 16-31 hold K {8..15,24..31}.
__device__ __forceinline__ v16bf load_frag(const __bf16* base, int stride,
                                           int row0, int k0, int lane) {
  int m  = lane & 15;
  int kb = (lane & 16) ? 8 : 0;
  const __bf16* p = base + (size_t)(row0 + m) * stride + k0 + kb;
  union { v16bf v; v8bf h[2]; } u;
  u.h[0] = *(const v8bf*)p;          // K = kb .. kb+7
  u.h[1] = *(const v8bf*)(p + 16);   // K = kb+16 .. kb+23
  return u.v;
}

// Store f32 accumulator tile (C/D layout) to a row-major bf16 buffer.
__device__ __forceinline__ void store_rm(__bf16* dst, int stride, int row0,
                                         int col, const v8f& acc, int M0) {
#pragma unroll
  for (int e = 0; e < 8; ++e)
    dst[(size_t)(row0 + M0 + e) * stride + col] = (__bf16)acc[e];
}

__device__ __forceinline__ void layernorm_to_a(Smem& S, const float* __restrict__ g,
                                               const float* __restrict__ bb, int tid) {
  int token = tid >> 2;      // 4 threads per token (quad within a wave)
  int q4    = tid & 3;
  float s = 0.f, s2 = 0.f;
  for (int j = q4; j < kC; j += 4) { float v = S.t[token][j]; s += v; s2 += v * v; }
  s  += __shfl_xor(s, 1);  s  += __shfl_xor(s, 2);
  s2 += __shfl_xor(s2, 1); s2 += __shfl_xor(s2, 2);
  float mean = s * (1.f / kC);
  float var  = s2 * (1.f / kC) - mean * mean;
  float rstd = rsqrtf(var + 1e-5f);
  for (int j = q4; j < kC; j += 4) {
    float v = (S.t[token][j] - mean) * rstd * g[j] + bb[j];
    S.A.a[token][j] = (__bf16)v;
  }
}

// One-time per launch: convert + transpose all weights to bf16 in workspace.
__global__ void prep_weights(const float* __restrict__ wqkv, const float* __restrict__ wproj,
                             const float* __restrict__ wm1,  const float* __restrict__ wm2,
                             __bf16* __restrict__ ws) {
  const int S0 = 576 * 192, S1 = 192 * 192, S2 = 768 * 192, S3 = 192 * 768;
  int idx = blockIdx.x * 256 + threadIdx.x;
  if (idx < S0) {                                   // wqkvT[576][192]
    int n = idx / 192, k = idx % 192;
    ws[idx] = (__bf16)wqkv[k * 576 + n];
  } else if (idx < S0 + S1) {                       // wprojT[192][192]
    int i = idx - S0; int n = i / 192, k = i % 192;
    ws[idx] = (__bf16)wproj[k * 192 + n];
  } else if (idx < S0 + S1 + S2) {                  // wm1T[768][192]
    int i = idx - S0 - S1; int n = i / 192, k = i % 192;
    ws[idx] = (__bf16)wm1[k * 768 + n];
  } else if (idx < S0 + S1 + S2 + S3) {             // wm2T[192][768]
    int i = idx - S0 - S1 - S2; int n = i / 768, k = i % 768;
    ws[idx] = (__bf16)wm2[k * 192 + n];
  }
}

__global__ __launch_bounds__(256, 1)
void swin_block_kernel(const float* __restrict__ x,
                       const float* __restrict__ b_qkv, const float* __restrict__ b_proj,
                       const float* __restrict__ ln1g, const float* __restrict__ ln1b,
                       const float* __restrict__ ln2g, const float* __restrict__ ln2b,
                       const float* __restrict__ b_m1, const float* __restrict__ b_m2,
                       const __bf16* __restrict__ wqkvT, const __bf16* __restrict__ wprojT,
                       const __bf16* __restrict__ wm1T, const __bf16* __restrict__ wm2T,
                       float* __restrict__ out) {
  extern __shared__ char smem_raw[];
  Smem& S = *reinterpret_cast<Smem*>(smem_raw);

  const int tid  = threadIdx.x;
  const int wave = tid >> 5;
  const int lane = tid & 31;
  const int ln16 = lane & 15;
  const int M0   = (lane & 16) ? 8 : 0;

  const int win = blockIdx.x;
  const int b   = win >> 10;
  const int wh  = (win >> 5) & 31;
  const int ww  = win & 31;
  const int h0  = wh * kWS, w0 = ww * kWS;

  // ---- Phase 0: gather window (B,C,H,W) -> t[64][192] f32 ----
  for (int idx = tid; idx < kN * kC; idx += 256) {
    int c = idx >> 6, n = idx & 63;
    int r = n >> 3, cw = n & 7;
    S.t[n][c] = x[(((size_t)b * kC + c) * kIMG + h0 + r) * kIMG + w0 + cw];
  }
  __syncthreads();

  // ---- Phase 1: LN1 -> A.a (bf16) ----
  layernorm_to_a(S, ln1g, ln1b, tid);
  __syncthreads();

  // ---- Phase 2: QKV = a @ WqkvT'  (64x576, 144 tiles, K=192) ----
  for (int tile = wave; tile < 144; tile += 8) {
    int mt = tile & 3, nt = tile >> 2;       // nt 0..35
    v8f acc = {};
#pragma unroll
    for (int kk = 0; kk < 6; ++kk) {
      v16bf af = load_frag(&S.A.a[0][0], SA, mt * 16, kk * 32, lane);
      v16bf bf = load_frag(wqkvT, kC, nt * 16, kk * 32, lane);
      acc = wmma_bf16(af, bf, acc);
    }
    int col = nt * 16 + ln16;
    float bias = b_qkv[col];
#pragma unroll
    for (int e = 0; e < 8; ++e) acc[e] += bias;
    if (nt < 12) {
      store_rm(&S.B.qkv.q[0][0], SQK, mt * 16, col, acc, M0);
    } else if (nt < 24) {
      store_rm(&S.B.qkv.k[0][0], SQK, mt * 16, col - kC, acc, M0);
    } else {                                  // V stored transposed: vT[channel][token]
      int c = col - 2 * kC;
      __bf16* p = &S.B.qkv.vT[c][mt * 16 + M0];
#pragma unroll
      for (int e = 0; e < 8; ++e) p[e] = (__bf16)acc[e];
    }
  }
  __syncthreads();

  // ---- Phase 3a: scores = (q kT) * scale  (6 heads x 16 tiles, single K=32 WMMA) ----
  const float scale = 0.17677669529663687f;   // 32^-0.5
  for (int tile = wave; tile < kHeads * 16; tile += 8) {
    int h = tile >> 4, tt = tile & 15;
    int mt = tt & 3, nt = tt >> 2;
    v16bf qf = load_frag(&S.B.qkv.q[0][0], SQK, mt * 16, h * kD, lane);
    v16bf kf = load_frag(&S.B.qkv.k[0][0], SQK, nt * 16, h * kD, lane);
    v8f acc = {};
    acc = wmma_bf16(qf, kf, acc);
    int col = nt * 16 + ln16;
#pragma unroll
    for (int e = 0; e < 8; ++e)
      S.A.sc[h][mt * 16 + M0 + e][col] = (__bf16)(acc[e] * scale);
  }
  __syncthreads();

  // ---- Phase 3b: softmax over 64-wide rows (one row per thread) ----
  for (int r = tid; r < kHeads * kN; r += 256) {
    int h = r >> 6, m = r & 63;
    __bf16* row = S.A.sc[h][m];
    float mx = -1e30f;
    for (int j = 0; j < kN; ++j) mx = fmaxf(mx, (float)row[j]);
    float sum = 0.f;
    for (int j = 0; j < kN; ++j) {
      float e = __expf((float)row[j] - mx);
      sum += e;
      row[j] = (__bf16)e;
    }
    float inv = 1.f / sum;
    for (int j = 0; j < kN; ++j) row[j] = (__bf16)((float)row[j] * inv);
  }
  __syncthreads();

  // ---- Phase 3c: o = probs @ v  (writes over dead q buffer) ----
  __bf16* o = &S.B.qkv.q[0][0];               // 64 x 192, stride SQK
  for (int tile = wave; tile < kHeads * 8; tile += 8) {
    int h = tile >> 3, tt = tile & 7;
    int mt = tt & 3, nt = tt >> 2;            // nt 0..1
    v8f acc = {};
#pragma unroll
    for (int kk = 0; kk < 2; ++kk) {
      v16bf pf = load_frag(&S.A.sc[h][0][0], SSC, mt * 16, kk * 32, lane);
      v16bf vf = load_frag(&S.B.qkv.vT[0][0], SVT, h * kD + nt * 16, kk * 32, lane);
      acc = wmma_bf16(pf, vf, acc);
    }
    store_rm(o, SQK, mt * 16, h * kD + nt * 16 + ln16, acc, M0);
  }
  __syncthreads();

  // ---- Phase 4: t += o @ WprojT' + b  (48 tiles, K=192) ----
  for (int tile = wave; tile < 48; tile += 8) {
    int mt = tile & 3, nt = tile >> 2;
    v8f acc = {};
#pragma unroll
    for (int kk = 0; kk < 6; ++kk) {
      v16bf af = load_frag(o, SQK, mt * 16, kk * 32, lane);
      v16bf bf = load_frag(wprojT, kC, nt * 16, kk * 32, lane);
      acc = wmma_bf16(af, bf, acc);
    }
    int col = nt * 16 + ln16;
    float bias = b_proj[col];
#pragma unroll
    for (int e = 0; e < 8; ++e)
      S.t[mt * 16 + M0 + e][col] += acc[e] + bias;
  }
  __syncthreads();

  // ---- Phase 5: LN2 -> A.a ----
  layernorm_to_a(S, ln2g, ln2b, tid);
  __syncthreads();

  // ---- Phase 6: MLP in 2 K-chunks of 384; out accumulators stay in VGPRs ----
  v8f oacc[6];
  v8f zero = {};
#pragma unroll
  for (int j = 0; j < 6; ++j) oacc[j] = zero;

  for (int chunk = 0; chunk < 2; ++chunk) {
    // hidden = gelu(a @ Wm1T'[:, chunk])   (96 tiles, K=192)
    for (int tile = wave; tile < 96; tile += 8) {
      int mt = tile & 3, ct = tile >> 2;     // ct 0..23
      v8f acc = {};
#pragma unroll
      for (int kk = 0; kk < 6; ++kk) {
        v16bf af = load_frag(&S.A.a[0][0], SA, mt * 16, kk * 32, lane);
        v16bf bf = load_frag(wm1T, kC, chunk * 384 + ct * 16, kk * 32, lane);
        acc = wmma_bf16(af, bf, acc);
      }
      int colg = chunk * 384 + ct * 16 + ln16;
      float bias = b_m1[colg];
#pragma unroll
      for (int e = 0; e < 8; ++e) {
        float v = acc[e] + bias;
        v = 0.5f * v * (1.f + erff(v * 0.70710678118654752f));   // exact GELU
        S.B.hid[mt * 16 + M0 + e][ct * 16 + ln16] = (__bf16)v;
      }
    }
    __syncthreads();
    // out += hidden @ Wm2T'[chunk]   (6 tiles per wave, K=384)
#pragma unroll
    for (int j = 0; j < 6; ++j) {
      int tile = wave + 8 * j;
      int mt = tile & 3, nt = tile >> 2;
      v8f acc = oacc[j];
      for (int kk = 0; kk < 12; ++kk) {
        v16bf af = load_frag(&S.B.hid[0][0], SHID, mt * 16, kk * 32, lane);
        v16bf bf = load_frag(wm2T, kMLP, nt * 16, chunk * 384 + kk * 32, lane);
        acc = wmma_bf16(af, bf, acc);
      }
      oacc[j] = acc;
    }
    __syncthreads();   // hid consumed before next chunk overwrites it
  }

  // ---- Phase 7: final residual + scatter back to (B,C,H,W) ----
#pragma unroll
  for (int j = 0; j < 6; ++j) {
    int tile = wave + 8 * j;
    int mt = tile & 3, nt = tile >> 2;
    int col = nt * 16 + ln16;
    float bias = b_m2[col];
    int rowbase = mt * 16 + M0;              // multiple of 8 -> 8 contiguous w
    int r = rowbase >> 3;
    float buf[8];
#pragma unroll
    for (int e = 0; e < 8; ++e)
      buf[e] = S.t[rowbase + e][col] + oacc[j][e] + bias;
    float* gp = out + (((size_t)b * kC + col) * kIMG + h0 + r) * kIMG + w0;
    *(float4*)gp       = make_float4(buf[0], buf[1], buf[2], buf[3]);
    *((float4*)gp + 1) = make_float4(buf[4], buf[5], buf[6], buf[7]);
  }
}

extern "C" void kernel_launch(void* const* d_in, const int* in_sizes, int n_in,
                              void* d_out, int out_size, void* d_ws, size_t ws_size,
                              hipStream_t stream) {
  (void)in_sizes; (void)n_in; (void)out_size; (void)ws_size;
  const float* x     = (const float*)d_in[0];
  const float* wqkv  = (const float*)d_in[1];
  const float* bqkv  = (const float*)d_in[2];
  const float* wproj = (const float*)d_in[3];
  const float* bproj = (const float*)d_in[4];
  const float* g1    = (const float*)d_in[5];
  const float* b1    = (const float*)d_in[6];
  const float* g2    = (const float*)d_in[7];
  const float* b2    = (const float*)d_in[8];
  const float* wm1   = (const float*)d_in[9];
  const float* bm1   = (const float*)d_in[10];
  const float* wm2   = (const float*)d_in[11];
  const float* bm2   = (const float*)d_in[12];

  __bf16* ws = (__bf16*)d_ws;
  const __bf16* qkvT  = ws;
  const __bf16* projT = qkvT + 576 * 192;
  const __bf16* m1T   = projT + 192 * 192;
  const __bf16* m2T   = m1T + 768 * 192;

  int total = 576 * 192 + 192 * 192 + 768 * 192 + 192 * 768;
  prep_weights<<<(total + 255) / 256, 256, 0, stream>>>(wqkv, wproj, wm1, wm2, ws);

  swin_block_kernel<<<kNWin, 256, sizeof(Smem), stream>>>(
      x, bqkv, bproj, g1, b1, g2, b2, bm1, bm2, qkvT, projT, m1T, m2T, (float*)d_out);
}
